// CoC_Conv_69526930587659
// MI455X (gfx1250) — matrix operand order repaired and checked by reading, not compiled, not run on hardware
//
#include <hip/hip_runtime.h>
#include <hip/hip_bf16.h>
#include <math.h>

// ---------------- problem constants (match reference) ----------------
constexpr int Bsz = 16, Cch = 256, HH = 64, WW = 64, HW = 4096;
constexpr int HEADS = 4, HEAD_DIM = 24, ED = 96, OUT_CH = 256, HID = 1024;
constexpr float GN_EPS = 1e-5f, BN_EPS = 1e-3f;

typedef __attribute__((ext_vector_type(16))) _Float16 v16h;
typedef __attribute__((ext_vector_type(8)))  _Float16 v8h;
typedef __attribute__((ext_vector_type(8)))  float    v8f;
typedef __attribute__((ext_vector_type(4)))  int      v4i;

// ---- CDNA5 async global->LDS copy (probe-guarded, safe fallback) ----
#if defined(__has_builtin)
#if __has_builtin(__builtin_amdgcn_global_load_async_to_lds_b128)
#define HAVE_ASYNC_COPY 1
#else
#define HAVE_ASYNC_COPY 0
#endif
#if __has_builtin(__builtin_amdgcn_s_wait_asynccnt)
#define HAVE_WAIT_ASYNC 1
#else
#define HAVE_WAIT_ASYNC 0
#endif
#else
#define HAVE_ASYNC_COPY 0
#define HAVE_WAIT_ASYNC 0
#endif

__device__ __forceinline__ void async_copy16(const _Float16* g, _Float16* l)
{
#if HAVE_ASYNC_COPY
    // toolchain signature: (v4i addrspace(1)*, v4i addrspace(3)*, imm offset, imm cpol)
    __builtin_amdgcn_global_load_async_to_lds_b128(
        (__attribute__((address_space(1))) v4i*)g,
        (__attribute__((address_space(3))) v4i*)l, 0, 0);
#else
    *(v8h*)l = *(const v8h*)g;   // sync fallback: reg-staged ds_store_b128
#endif
}

// wait for this wave's outstanding async copies, then workgroup barrier
__device__ __forceinline__ void async_stage_wait()
{
#if HAVE_ASYNC_COPY
#if HAVE_WAIT_ASYNC
    __builtin_amdgcn_s_wait_asynccnt(0);
#else
    asm volatile("s_wait_asynccnt 0" ::: "memory");
#endif
#endif
    __syncthreads();
}

// ======================================================================
// GroupNorm (per-batch over C*H*W) statistics
// ======================================================================
__global__ __launch_bounds__(256)
void gn_stats_kernel(const float* __restrict__ x, float* __restrict__ stats, int perb)
{
    __shared__ float sh1[256];
    __shared__ float sh2[256];
    const int b = blockIdx.x, tid = threadIdx.x;
    const float* p = x + (size_t)b * perb;
    float s = 0.f, s2 = 0.f;
    for (int i = tid; i < perb; i += 256) { float v = p[i]; s += v; s2 += v * v; }
    sh1[tid] = s; sh2[tid] = s2;
    __syncthreads();
    for (int off = 128; off > 0; off >>= 1) {
        if (tid < off) { sh1[tid] += sh1[tid + off]; sh2[tid] += sh2[tid + off]; }
        __syncthreads();
    }
    if (tid == 0) {
        float inv = 1.f / (float)perb;
        float mean = sh1[0] * inv;
        float var  = sh2[0] * inv - mean * mean;
        stats[2 * b]     = mean;
        stats[2 * b + 1] = rsqrtf(var + GN_EPS);
    }
}

// x is NCHW fp32 -> normalized f16 in transposed [B][HW][C] (channel-fastest)
__global__ __launch_bounds__(256)
void gn_apply_nchw_to_t(const float* __restrict__ x, const float* __restrict__ stats,
                        const float* __restrict__ g, const float* __restrict__ bt,
                        _Float16* __restrict__ dst)
{
    size_t i = (size_t)blockIdx.x * 256 + threadIdx.x;
    if (i >= (size_t)Bsz * Cch * HW) return;
    int p = (int)(i & (HW - 1));
    size_t t = i >> 12;            // / HW
    int c = (int)(t & (Cch - 1));
    int b = (int)(t >> 8);         // / C
    float v = (x[i] - stats[2 * b]) * stats[2 * b + 1] * g[c] + bt[c];
    dst[((size_t)b * HW + p) * Cch + c] = (_Float16)v;
}

// y already in [B][HW][C] fp32 -> normalized f16 same layout (fully coalesced)
__global__ __launch_bounds__(256)
void gn_apply_t_to_t(const float* __restrict__ y, const float* __restrict__ stats,
                     const float* __restrict__ g, const float* __restrict__ bt,
                     _Float16* __restrict__ dst)
{
    size_t i = (size_t)blockIdx.x * 256 + threadIdx.x;
    if (i >= (size_t)Bsz * HW * Cch) return;
    int c = (int)(i & (Cch - 1));
    int b = (int)(i >> 20);        // / (HW*C)
    float v = (y[i] - stats[2 * b]) * stats[2 * b + 1] * g[c] + bt[c];
    dst[i] = (_Float16)v;
}

__global__ __launch_bounds__(256)
void cvt_f32_to_f16(const float* __restrict__ src, _Float16* __restrict__ dst, int n)
{
    int i = blockIdx.x * 256 + threadIdx.x;
    if (i < n) dst[i] = (_Float16)src[i];
}

// ======================================================================
// WMMA GEMM:  D[b](M,N) = Wh(M,K) * Xt[b](K,N) (+fused epilogue)
//   Wh : f16 [M][K] row-major (streamed from global, L2-resident)
//   Xt : f16 [B][N][K] channel-fastest activations
//   4-wave workgroup: shared 64(N) x 32(K) B tile double-buffered in LDS
//   via async global->LDS copies; wave w computes M-tile blockIdx.y*64+w*16.
// ======================================================================
enum {
    EPI_BIAS_F32T = 0,        // out f32 [B][N][M] = acc + bias
    EPI_RES1_F32T = 1,        // y = x(NCHW) + ls1[m]*(acc+bias), f32 [B][N][M]
    EPI_GELU_F16T = 2,        // gelu(acc+bias), f16 [B][N][M]
    EPI_RES2_BN_RELU_F16T = 3,// y2=y+ls2*(acc+bias); relu(BN(y2*dw_w)), f16 [B][N][M]
    EPI_BN_RELU_NCHW = 4      // relu(BN(acc)) -> f32 NCHW (final output)
};

struct EpiArgs {
    const float* a0; const float* a1; const float* a2; const float* a3;
    const float* a4; const float* a5; const float* a6;
};

__device__ inline float gelu_tanh(float x)
{
    float x3 = x * x * x;
    return 0.5f * x * (1.0f + tanhf(0.7978845608028654f * (x + 0.044715f * x3)));
}

template<int EPI>
__global__ __launch_bounds__(128)
void wmma_gemm_kernel(const _Float16* __restrict__ Wh,
                      const _Float16* __restrict__ Xt,
                      const float* __restrict__ bias,
                      float* __restrict__ outF, _Float16* __restrict__ outH,
                      int M, int N, int K, EpiArgs ep)
{
    __shared__ _Float16 sB[2][64 * 32];            // double-buffered B tile, 2x4KB

    const int tid  = threadIdx.x;                  // 0..127 (4 waves)
    const int wave = tid >> 5;
    const int lane = tid & 31;
    const int lm = lane & 15;                      // M index (A), N index (B/D)
    const int lh = lane >> 4;                      // lane-half selector
    const int n0 = blockIdx.x * 64;
    const int b  = blockIdx.z;

    const int  m0raw  = blockIdx.y * 64 + wave * 16;
    const bool mvalid = (m0raw < M);               // idle waves still barrier/stage
    const int  m0     = mvalid ? m0raw : 0;

    const _Float16* __restrict__ Xb = Xt + (size_t)b * N * K;
    // A row for this lane: elems 0..7 -> k0+lh*8.., elems 8..15 -> k0+16+lh*8..
    const _Float16* __restrict__ arow = Wh + (size_t)(m0 + lm) * K + lh * 8;

    // stage the 64x32 B tile at column k0 into LDS buffer `buf`
    auto stage = [&](int k0, int buf) {
        #pragma unroll
        for (int r = 0; r < 2; ++r) {
            int idx = tid + r * 128;               // 256 x 16B segments
            int row = idx >> 2, seg = idx & 3;
            const _Float16* g = Xb + (size_t)(n0 + row) * K + k0 + seg * 8;
            async_copy16(g, &sB[buf][row * 32 + seg * 8]);
        }
    };

    v8f acc[4] = {};
    const int nk = K / 32;
    stage(0, 0);

    for (int kk = 0; kk < nk; ++kk) {
        // sB[kk&1] ready for all waves; previous reads of the other buffer done
        async_stage_wait();
        if (kk + 1 < nk) stage((kk + 1) * 32, (kk + 1) & 1);

        const int k0 = kk * 32;
        if (kk + 1 < nk) __builtin_prefetch(arow + k0 + 32, 0, 1); // global_prefetch_b8
        v8h alo = *(const v8h*)(arow + k0);
        v8h ahi = *(const v8h*)(arow + k0 + 16);
        v16h afrag = __builtin_shufflevector(alo, ahi,
                        0,1,2,3,4,5,6,7,8,9,10,11,12,13,14,15);
        const _Float16* __restrict__ bbuf = sB[kk & 1];
        #pragma unroll
        for (int j = 0; j < 4; ++j) {
            // lane holds column n = j*16+lm; K halves lh*8..+7 and 16+lh*8..+7
            const _Float16* brow = bbuf + (j * 16 + lm) * 32 + lh * 8;
            v8h blo = *(const v8h*)(brow);          // ds_load_b128
            v8h bhi = *(const v8h*)(brow + 16);
            v16h bfrag = __builtin_shufflevector(blo, bhi,
                            0,1,2,3,4,5,6,7,8,9,10,11,12,13,14,15);
            acc[j] = __builtin_amdgcn_wmma_f32_16x16x32_f16(
                         false, afrag, false, bfrag, (short)0, acc[j], false, false);
        }
    }

    if (!mvalid) return;

    // Epilogue. D layout: lane -> n = n0+j*16+lm ; element v -> m = m0+lh*8+v
    const int mbase = m0 + lh * 8;
    #pragma unroll
    for (int j = 0; j < 4; ++j) {
        const int n = n0 + j * 16 + lm;
        if constexpr (EPI == EPI_BIAS_F32T) {
            float* dst = outF + ((size_t)b * N + n) * M + mbase;
            #pragma unroll
            for (int v = 0; v < 8; ++v) dst[v] = acc[j][v] + bias[mbase + v];
        } else if constexpr (EPI == EPI_RES1_F32T) {
            // a0 = x (NCHW), a1 = ls1
            float* dst = outF + ((size_t)b * N + n) * M + mbase;
            #pragma unroll
            for (int v = 0; v < 8; ++v) {
                int m = mbase + v;
                float g = acc[j][v] + bias[m];
                dst[v] = ep.a0[((size_t)b * M + m) * N + n] + ep.a1[m] * g;
            }
        } else if constexpr (EPI == EPI_GELU_F16T) {
            _Float16* dst = outH + ((size_t)b * N + n) * M + mbase;
            #pragma unroll
            for (int v = 0; v < 8; ++v)
                dst[v] = (_Float16)gelu_tanh(acc[j][v] + bias[mbase + v]);
        } else if constexpr (EPI == EPI_RES2_BN_RELU_F16T) {
            // a0=y_t [B][N][M], a1=ls2, a2=dw_w, a3=dw_g, a4=dw_b, a5=dw_m, a6=dw_v
            const float* yrow = ep.a0 + ((size_t)b * N + n) * M + mbase;
            _Float16* dst = outH + ((size_t)b * N + n) * M + mbase;
            #pragma unroll
            for (int v = 0; v < 8; ++v) {
                int m = mbase + v;
                float g  = acc[j][v] + bias[m];
                float y2 = yrow[v] + ep.a1[m] * g;
                float t  = y2 * ep.a2[m];
                float s  = ep.a3[m] * rsqrtf(ep.a6[m] + BN_EPS);
                float z  = (t - ep.a5[m]) * s + ep.a4[m];
                dst[v] = (_Float16)fmaxf(z, 0.f);
            }
        } else { // EPI_BN_RELU_NCHW: a0=pw_g, a1=pw_b, a2=pw_m, a3=pw_v
            #pragma unroll
            for (int v = 0; v < 8; ++v) {
                int m = mbase + v;
                float s = ep.a0[m] * rsqrtf(ep.a3[m] + BN_EPS);
                float z = (acc[j][v] - ep.a2[m]) * s + ep.a1[m];
                outF[((size_t)b * M + m) * N + n] = fmaxf(z, 0.f);
            }
        }
    }
}

// ======================================================================
// Context-cluster core: 256 sub-batches, each 1024 points x 24ch, 4 centers
// feat_t/val_t: f32 [B][HW][ED];  co_h: f16 [B][HW][ED]
// ======================================================================
__global__ __launch_bounds__(256)
void cluster_kernel(const float* __restrict__ feat_t, const float* __restrict__ val_t,
                    const float* __restrict__ alphaP, const float* __restrict__ betaP,
                    _Float16* __restrict__ co_h)
{
    __shared__ float s_cen[4][24];
    __shared__ float s_vcen[4][24];
    __shared__ float s_agg[4][24];
    __shared__ float s_cnt[4];
    __shared__ float s_cinv[4];
    __shared__ float s_sim[1024];
    __shared__ int   s_idx[1024];

    const int tid = threadIdx.x;
    const int bp = blockIdx.x;          // b' = ((b*4+e)*2+f1)*2+f2
    const int b  = bp >> 4;
    const int e  = (bp >> 2) & 3;
    const int f1 = (bp >> 1) & 1;
    const int f2 = bp & 1;
    const float alpha = alphaP[0], beta = betaP[0];

    const size_t baseB = (size_t)b * HW * ED;
    const int ch0 = e * HEAD_DIM;

    // ---- pooled centers (mean over each 16x16 quadrant of the 32x32 tile) ----
    if (tid < 96) {
        int m = tid / 24, c = tid % 24;
        int p1 = m >> 1, p2 = m & 1;
        float sf = 0.f, sv = 0.f;
        for (int i = 0; i < 256; ++i) {
            int h = p1 * 16 + (i >> 4);
            int w = p2 * 16 + (i & 15);
            int hw = (f1 * 32 + h) * WW + (f2 * 32 + w);
            size_t off = baseB + (size_t)hw * ED + ch0 + c;
            sf += feat_t[off];
            sv += val_t[off];
        }
        s_cen[m][c]  = sf * (1.f / 256.f);
        s_vcen[m][c] = sv * (1.f / 256.f);
        s_agg[m][c]  = 0.f;
    }
    if (tid < 4) s_cnt[tid] = 0.f;
    __syncthreads();
    if (tid < 4) {
        float s = 0.f;
        for (int c = 0; c < 24; ++c) { float v = s_cen[tid][c]; s += v * v; }
        s_cinv[tid] = rsqrtf(s + 1e-12f);
    }
    __syncthreads();

    // ---- per-point similarity, hard assignment, LDS-atomic aggregation ----
    for (int n = tid; n < 1024; n += 256) {
        int h = n >> 5, w = n & 31;
        int hw = (f1 * 32 + h) * WW + (f2 * 32 + w);
        size_t off = baseB + (size_t)hw * ED + ch0;
        float pt[24];
        float ss = 0.f;
        #pragma unroll
        for (int c = 0; c < 24; ++c) { pt[c] = feat_t[off + c]; ss += pt[c] * pt[c]; }
        float pinv = rsqrtf(ss + 1e-12f);
        float best = -1.f; int bi = 0;
        #pragma unroll
        for (int m = 0; m < 4; ++m) {
            float d = 0.f;
            #pragma unroll
            for (int c = 0; c < 24; ++c) d += s_cen[m][c] * pt[c];
            float sim = 1.f / (1.f + expf(-(beta + alpha * d * s_cinv[m] * pinv)));
            if (sim > best) { best = sim; bi = m; }   // first-max like jnp.argmax
        }
        s_sim[n] = best; s_idx[n] = bi;
        #pragma unroll
        for (int c = 0; c < 24; ++c)
            atomicAdd(&s_agg[bi][c], best * val_t[off + c]);   // ds_add_f32
        atomicAdd(&s_cnt[bi], 1.f);
    }
    __syncthreads();
    if (tid < 96) {
        int m = tid / 24, c = tid % 24;
        s_agg[m][c] = (s_agg[m][c] + s_vcen[m][c]) / (s_cnt[m] + 1.f);
    }
    __syncthreads();

    // ---- dispatch back to points, straight into f16 [N][K] for proj GEMM ----
    for (int n = tid; n < 1024; n += 256) {
        int h = n >> 5, w = n & 31;
        int hw = (f1 * 32 + h) * WW + (f2 * 32 + w);
        size_t off = baseB + (size_t)hw * ED + ch0;
        float sm = s_sim[n]; int m = s_idx[n];
        #pragma unroll
        for (int c = 0; c < 24; ++c)
            co_h[off + c] = (_Float16)(sm * s_agg[m][c]);
    }
}

// ======================================================================
// host launch
// ======================================================================
extern "C" void kernel_launch(void* const* d_in, const int* in_sizes, int n_in,
                              void* d_out, int out_size, void* d_ws, size_t ws_size,
                              hipStream_t stream)
{
    (void)in_sizes; (void)n_in; (void)out_size; (void)ws_size;
    const float* x      = (const float*)d_in[0];
    const float* gn1_w  = (const float*)d_in[1];
    const float* gn1_b  = (const float*)d_in[2];
    const float* f_w    = (const float*)d_in[3];
    const float* f_b    = (const float*)d_in[4];
    const float* v_w    = (const float*)d_in[5];
    const float* v_b    = (const float*)d_in[6];
    const float* proj_w = (const float*)d_in[7];
    const float* proj_b = (const float*)d_in[8];
    const float* sim_a  = (const float*)d_in[9];
    const float* sim_b  = (const float*)d_in[10];
    const float* ls1    = (const float*)d_in[11];
    const float* gn2_w  = (const float*)d_in[12];
    const float* gn2_b  = (const float*)d_in[13];
    const float* mlp_w1 = (const float*)d_in[14];
    const float* mlp_b1 = (const float*)d_in[15];
    const float* mlp_w2 = (const float*)d_in[16];
    const float* mlp_b2 = (const float*)d_in[17];
    const float* ls2    = (const float*)d_in[18];
    const float* dw_w   = (const float*)d_in[19];
    const float* dw_g   = (const float*)d_in[20];
    const float* dw_b   = (const float*)d_in[21];
    const float* dw_m   = (const float*)d_in[22];
    const float* dw_v   = (const float*)d_in[23];
    const float* pw_w   = (const float*)d_in[24];
    const float* pw_g   = (const float*)d_in[25];
    const float* pw_b   = (const float*)d_in[26];
    const float* pw_m   = (const float*)d_in[27];
    const float* pw_v   = (const float*)d_in[28];
    float* out = (float*)d_out;

    // ---- workspace carve (deterministic, 256B aligned) ----
    char* wsp = (char*)d_ws;
    auto carve = [&](size_t bytes) -> char* {
        char* p = wsp; wsp += (bytes + 255) & ~(size_t)255; return p;
    };
    float*    stats1 = (float*)carve(Bsz * 2 * sizeof(float));
    float*    stats2 = (float*)carve(Bsz * 2 * sizeof(float));
    _Float16* wh_f   = (_Float16*)carve((size_t)ED * Cch * 2);
    _Float16* wh_v   = (_Float16*)carve((size_t)ED * Cch * 2);
    _Float16* wh_pj  = (_Float16*)carve((size_t)Cch * ED * 2);
    _Float16* wh_m1  = (_Float16*)carve((size_t)HID * Cch * 2);
    _Float16* wh_m2  = (_Float16*)carve((size_t)Cch * HID * 2);
    _Float16* wh_pw  = (_Float16*)carve((size_t)OUT_CH * Cch * 2);
    _Float16* xn_h   = (_Float16*)carve((size_t)Bsz * HW * Cch * 2);  // reused as yn_h
    float*    feat_t = (float*)carve((size_t)Bsz * HW * ED * 4);      // feat+val reused as z_h
    float*    val_t  = (float*)carve((size_t)Bsz * HW * ED * 4);
    _Float16* co_h   = (_Float16*)carve((size_t)Bsz * HW * ED * 2);
    float*    y_t    = (float*)carve((size_t)Bsz * HW * Cch * 4);
    _Float16* hdn_h  = (_Float16*)carve((size_t)Bsz * HW * HID * 2);
    _Float16* yn_h   = xn_h;                 // xn dead after feat/val GEMMs
    _Float16* z_h    = (_Float16*)feat_t;    // feat/val dead after cluster

    const size_t totalA = (size_t)Bsz * Cch * HW;
    EpiArgs ep0{};

    // 1) GN1 -> normalized f16 activations in [B][N][K]
    gn_stats_kernel<<<Bsz, 256, 0, stream>>>(x, stats1, Cch * HW);
    gn_apply_nchw_to_t<<<(unsigned)((totalA + 255) / 256), 256, 0, stream>>>(
        x, stats1, gn1_w, gn1_b, xn_h);

    // 2) weight converts
    cvt_f32_to_f16<<<(ED * Cch + 255) / 256, 256, 0, stream>>>(f_w, wh_f, ED * Cch);
    cvt_f32_to_f16<<<(ED * Cch + 255) / 256, 256, 0, stream>>>(v_w, wh_v, ED * Cch);
    cvt_f32_to_f16<<<(Cch * ED + 255) / 256, 256, 0, stream>>>(proj_w, wh_pj, Cch * ED);
    cvt_f32_to_f16<<<(HID * Cch + 255) / 256, 256, 0, stream>>>(mlp_w1, wh_m1, HID * Cch);
    cvt_f32_to_f16<<<(Cch * HID + 255) / 256, 256, 0, stream>>>(mlp_w2, wh_m2, Cch * HID);
    cvt_f32_to_f16<<<(OUT_CH * Cch + 255) / 256, 256, 0, stream>>>(pw_w, wh_pw, OUT_CH * Cch);

    // 3) feat / val 1x1 convs (WMMA + async-LDS staging)
    wmma_gemm_kernel<EPI_BIAS_F32T><<<dim3(HW / 64, (ED + 63) / 64, Bsz), 128, 0, stream>>>(
        wh_f, xn_h, f_b, feat_t, nullptr, ED, HW, Cch, ep0);
    wmma_gemm_kernel<EPI_BIAS_F32T><<<dim3(HW / 64, (ED + 63) / 64, Bsz), 128, 0, stream>>>(
        wh_v, xn_h, v_b, val_t, nullptr, ED, HW, Cch, ep0);

    // 4) clustering core (LDS atomics)
    cluster_kernel<<<256, 256, 0, stream>>>(feat_t, val_t, sim_a, sim_b, co_h);

    // 5) proj conv fused with residual + layerscale -> y
    {
        EpiArgs e1{}; e1.a0 = x; e1.a1 = ls1;
        wmma_gemm_kernel<EPI_RES1_F32T><<<dim3(HW / 64, Cch / 64, Bsz), 128, 0, stream>>>(
            wh_pj, co_h, proj_b, y_t, nullptr, Cch, HW, ED, e1);
    }

    // 6) GN2 -> f16 (same layout, coalesced)
    gn_stats_kernel<<<Bsz, 256, 0, stream>>>(y_t, stats2, Cch * HW);
    gn_apply_t_to_t<<<(unsigned)((totalA + 255) / 256), 256, 0, stream>>>(
        y_t, stats2, gn2_w, gn2_b, yn_h);

    // 7) MLP1 + GELU
    wmma_gemm_kernel<EPI_GELU_F16T><<<dim3(HW / 64, HID / 64, Bsz), 128, 0, stream>>>(
        wh_m1, yn_h, mlp_b1, nullptr, hdn_h, HID, HW, Cch, ep0);

    // 8) MLP2 fused with residual+ls2 and the depthwise(1x1)+BN+ReLU stage
    {
        EpiArgs e2{}; e2.a0 = y_t; e2.a1 = ls2; e2.a2 = dw_w;
        e2.a3 = dw_g; e2.a4 = dw_b; e2.a5 = dw_m; e2.a6 = dw_v;
        wmma_gemm_kernel<EPI_RES2_BN_RELU_F16T><<<dim3(HW / 64, Cch / 64, Bsz), 128, 0, stream>>>(
            wh_m2, hdn_h, mlp_b2, nullptr, z_h, Cch, HW, HID, e2);
    }

    // 9) pointwise conv + BN + ReLU -> final NCHW fp32 output
    {
        EpiArgs e3{}; e3.a0 = pw_g; e3.a1 = pw_b; e3.a2 = pw_m; e3.a3 = pw_v;
        wmma_gemm_kernel<EPI_BN_RELU_NCHW><<<dim3(HW / 64, OUT_CH / 64, Bsz), 128, 0, stream>>>(
            wh_pw, z_h, nullptr, out, nullptr, OUT_CH, HW, Cch, e3);
    }
}
